// ChunkBasedMultiHeadAttention_81183471829328
// MI455X (gfx1250) — matrix operand order repaired and checked by reading, not compiled, not run on hardware
//
#include <hip/hip_runtime.h>

// ---------------- problem constants (from reference) ----------------
#define S_LEN  2048
#define BATCH  2
#define EMB    1024
#define NH     16
#define HD     64          // head dim
#define CHUNK  64
#define M_ROWS (S_LEN * BATCH)   // 4096 flattened (s,b) rows
#define N_QKV  (3 * EMB)         // 3072

typedef _Float16 h8  __attribute__((ext_vector_type(8)));
typedef _Float16 h16 __attribute__((ext_vector_type(16)));
typedef float    f8  __attribute__((ext_vector_type(8)));

// ---------------- WMMA wrapper: D = A(16x32 f16) * B(32x16 f16) + C(f32) ----
__device__ __forceinline__ f8 wmma32(h16 a, h16 b, f8 c) {
  return __builtin_amdgcn_wmma_f32_16x16x32_f16(
      /*neg_a=*/false, a, /*neg_b=*/false, b,
      /*c_mod=*/(short)0, c, /*reuse_a=*/false, /*reuse_b=*/false);
}

// A-fragment loader (ISA 16-bit A 16x32 layout):
//   lane L: row = L&15, hs = L>>4
//   elements 0..7  = K (k0 + hs*8 + 0..7), elements 8..15 = K (k0 + 16 + hs*8 + 0..7)
__device__ __forceinline__ h16 frag_a(const _Float16* __restrict__ row, int k0, int hs) {
  h8 lo = *(const h8*)(row + k0 + hs * 8);
  h8 hi = *(const h8*)(row + k0 + 16 + hs * 8);
  return __builtin_shufflevector(lo, hi, 0,1,2,3,4,5,6,7,8,9,10,11,12,13,14,15);
}

// B-fragment loader (ISA 16-bit B 32x16 layout, source stored row-major [N,K]):
//   lane L: column n = L&15 (its own source row), elements 0..7 = K (k0 + hs*8 + 0..7), 8..15 = K (k0 + 16 + hs*8 + 0..7)
__device__ __forceinline__ h16 frag_b(const _Float16* __restrict__ row, int k0, int hs) {
  h8 lo = *(const h8*)(row + k0 + hs * 8);
  h8 hi = *(const h8*)(row + k0 + 16 + hs * 8);
  return __builtin_shufflevector(lo, hi, 0,1,2,3,4,5,6,7,8,9,10,11,12,13,14,15);
}

// ---- CDNA5 async copy: global -> LDS, 16B per lane, tracked by ASYNCcnt ----
__device__ __forceinline__ void async_g2l_b128(void* lds_dst, const void* gsrc) {
  unsigned l = (unsigned)(unsigned long long)lds_dst;  // LDS aperture: low 32 bits = LDS offset
  asm volatile("global_load_async_to_lds_b128 %0, %1, off"
               :: "v"(l), "v"(gsrc) : "memory");
}
__device__ __forceinline__ void wait_async0() {
#if __has_builtin(__builtin_amdgcn_s_wait_asynccnt)
  __builtin_amdgcn_s_wait_asynccnt(0);
#else
  asm volatile("s_wait_asynccnt 0x0" ::: "memory");
#endif
}

// ---------------- K1: f32 -> f16 convert ----------------
__global__ void cvt_f32_f16(const float* __restrict__ src, _Float16* __restrict__ dst, int n) {
  for (int i = blockIdx.x * blockDim.x + threadIdx.x; i < n; i += gridDim.x * blockDim.x)
    dst[i] = (_Float16)src[i];
}

// ==========================================================================
// Pipelined WMMA GEMM core: each wave computes a 32x64 tile of C = A * W^T.
// Ping-pong fragment register sets (no copies, no steady-state branches);
// last k-step peeled so the loop body is pure load/WMMA overlap.
// ==========================================================================
#define LOAD_SET(XA, XB, K)                                                    \
  XA[0] = frag_a(ar0, (K), hs); XA[1] = frag_a(ar1, (K), hs);                  \
  _Pragma("unroll")                                                            \
  for (int t = 0; t < 4; ++t) XB[t] = frag_b(wr[t], (K), hs);

#define WMMA_SET(XA, XB)                                                       \
  _Pragma("unroll")                                                            \
  for (int t = 0; t < 4; ++t) {                                                \
    acc[0][t] = wmma32(XA[0], XB[t], acc[0][t]);                               \
    acc[1][t] = wmma32(XA[1], XB[t], acc[1][t]);                               \
  }

#define GEMM_BODY(KDIM)                                                        \
  const int lane = threadIdx.x & 31;                                           \
  const int wave = threadIdx.x >> 5;                                           \
  const int ln = lane & 15, hs = lane >> 4;                                    \
  const int m0 = blockIdx.y * 128 + wave * 32;                                 \
  const int n0 = blockIdx.x * 64;                                              \
  const _Float16* ar0 = A + (size_t)(m0 + ln) * (KDIM);                        \
  const _Float16* ar1 = A + (size_t)(m0 + 16 + ln) * (KDIM);                   \
  const _Float16* wr[4];                                                       \
  _Pragma("unroll")                                                            \
  for (int t = 0; t < 4; ++t) wr[t] = W + (size_t)(n0 + t * 16 + ln) * (KDIM); \
  f8 acc[2][4] = {};                                                           \
  h16 pa[2], pb[4], qa[2], qb[4];                                              \
  LOAD_SET(pa, pb, 0)                                                          \
  int k0 = 0;                                                                  \
  for (; k0 + 64 < (KDIM); k0 += 64) {                                         \
    __builtin_prefetch(ar0 + k0 + 512, 0, 1);                                  \
    LOAD_SET(qa, qb, k0 + 32)                                                  \
    WMMA_SET(pa, pb)                                                           \
    LOAD_SET(pa, pb, k0 + 64)                                                  \
    WMMA_SET(qa, qb)                                                           \
  }                                                                            \
  LOAD_SET(qa, qb, k0 + 32)                                                    \
  WMMA_SET(pa, pb)                                                             \
  WMMA_SET(qa, qb)

// ---------------- K2: qkv = x * W_in^T + b ; scatter to q/k/vT head layouts --
__global__ void __launch_bounds__(128)
gemm_qkv(const _Float16* __restrict__ A,    // [M_ROWS, EMB]
         const _Float16* __restrict__ W,    // [3E, EMB] (row n has K contiguous)
         const float*    __restrict__ bias, // [3E] f32
         _Float16* __restrict__ qh,         // [B,NH,S,HD]  (pre-scaled by 1/8)
         _Float16* __restrict__ kh,         // [B,NH,S,HD]
         _Float16* __restrict__ vth)        // [B,NH,HD,S]  (transposed V)
{
  GEMM_BODY(EMB)
#pragma unroll
  for (int r = 0; r < 2; ++r) {
#pragma unroll
    for (int t = 0; t < 4; ++t) {
#pragma unroll
      for (int e = 0; e < 8; ++e) {
        const int m = m0 + r * 16 + hs * 8 + e;   // flattened row = s*BATCH + b
        const int n = n0 + t * 16 + ln;
        const int s = m >> 1, b = m & 1;          // BATCH == 2
        const float v = acc[r][t][e] + bias[n];
        if (n < EMB) {                            // Q (fold in 1/sqrt(64))
          const int h = n >> 6, d = n & 63;
          qh[(((size_t)(b * NH + h)) * S_LEN + s) * HD + d] = (_Float16)(v * 0.125f);
        } else if (n < 2 * EMB) {                 // K
          const int nn = n - EMB, h = nn >> 6, d = nn & 63;
          kh[(((size_t)(b * NH + h)) * S_LEN + s) * HD + d] = (_Float16)v;
        } else {                                  // V, stored transposed [D,S]
          const int nn = n - 2 * EMB, h = nn >> 6, d = nn & 63;
          vth[(((size_t)(b * NH + h)) * HD + d) * S_LEN + s] = (_Float16)v;
        }
      }
    }
  }
}

// ---------------- K4: out = attn * W_out^T + b (fp32 out) ----------------
__global__ void __launch_bounds__(128)
gemm_out(const _Float16* __restrict__ A,    // [M_ROWS, EMB]
         const _Float16* __restrict__ W,    // [EMB, EMB]
         const float*    __restrict__ bias, // [EMB]
         float* __restrict__ out)           // [M_ROWS, EMB] == [S,B,E]
{
  GEMM_BODY(EMB)
#pragma unroll
  for (int r = 0; r < 2; ++r)
#pragma unroll
    for (int t = 0; t < 4; ++t)
#pragma unroll
      for (int e = 0; e < 8; ++e) {
        const int m = m0 + r * 16 + hs * 8 + e;
        const int n = n0 + t * 16 + ln;
        out[(size_t)m * EMB + n] = acc[r][t][e] + bias[n];
      }
}

// ==========================================================================
// K3: streaming chunked attention. Chunk mask == causal over chunks + full
// bidirectional inside a chunk => iterating kc = 0..qc needs NO masking.
// K/V chunks are staged in double-buffered LDS via global_load_async_to_lds,
// shared by all 4 waves; next chunk's DMA overlaps current chunk's WMMAs.
// ==========================================================================
__global__ void __launch_bounds__(128)
attn_chunk(const _Float16* __restrict__ qh,   // [B,NH,S,HD], pre-scaled
           const _Float16* __restrict__ kh,   // [B,NH,S,HD]
           const _Float16* __restrict__ vth,  // [B,NH,HD,S]
           _Float16* __restrict__ attn)       // [S,B,EMB]
{
  __shared__ _Float16 kbuf[2][64 * 64];       // [t][d] rows, 8KB per buffer
  __shared__ _Float16 vbuf[2][64 * 64];       // [d][t] rows, 8KB per buffer
  __shared__ _Float16 pbuf[4][16 * 64];       // wave-private P staging

  const int tid  = threadIdx.x;
  const int lane = tid & 31;
  const int wave = tid >> 5;
  const int ln = lane & 15, hs = lane >> 4;
  const int qc = blockIdx.x;                  // query chunk 0..31
  const int bh = blockIdx.y;                  // b*NH + h
  const int b = bh >> 4, h = bh & 15;         // NH == 16

  const _Float16* qp = qh + (size_t)bh * S_LEN * HD;
  const _Float16* kp = kh + (size_t)bh * S_LEN * HD;
  const _Float16* vp = vth + (size_t)bh * HD * S_LEN;
  const int s0 = qc * CHUNK + wave * 16;      // this wave's 16 query rows

  const _Float16* qrow = qp + (size_t)(s0 + ln) * HD;
  const h16 qa0 = frag_a(qrow, 0, hs);
  const h16 qa1 = frag_a(qrow, 32, hs);

  f8 o[4] = {};
  float mrow[8], lsum[8];
#pragma unroll
  for (int e = 0; e < 8; ++e) { mrow[e] = -3.0e38f; lsum[e] = 0.0f; }

  _Float16* prow = &pbuf[wave][0];

  // ---- stage key/value chunk t0 into LDS buffer (8 x b128 async per thread)
  auto stage = [&](int buf, int t0) {
#pragma unroll
    for (int i = 0; i < 4; ++i) {
      const int u = i * 128 + tid;            // 0..511 16B units per matrix
      const int row = u >> 3, c = (u & 7) * 8;
      async_g2l_b128(&kbuf[buf][row * 64 + c], kp + (size_t)(t0 + row) * HD + c);
      async_g2l_b128(&vbuf[buf][row * 64 + c], vp + (size_t)row * S_LEN + t0 + c);
    }
  };

  stage(0, 0);
  for (int kc = 0; kc <= qc; ++kc) {
    const int cur = kc & 1;
    const int t0 = kc * CHUNK;
    wait_async0();        // my 8 loads for buffer `cur` have landed
    __syncthreads();      // everyone's landed; everyone done reading buffer cur^1
    if (kc < qc) stage(cur ^ 1, t0 + CHUNK);  // overlap next DMA with compute

    const _Float16* kb = &kbuf[cur][0];
    const _Float16* vb = &vbuf[cur][0];

    // S = Q * K^T  (16 x 64 strip, K-dim = HD = 64) -- 8 WMMAs from LDS
    f8 sc[4] = {};
#pragma unroll
    for (int half = 0; half < 2; ++half) {
      const h16 a = half ? qa1 : qa0;
      const int dt = half * 32;
#pragma unroll
      for (int t = 0; t < 4; ++t) {
        h16 kf = frag_b(kb + (t * 16 + ln) * 64, dt, hs);
        sc[t] = wmma32(a, kf, sc[t]);
      }
    }

    // online softmax: row max across 4 tiles + the 16 lanes of this half-wave
    float alpha[8], rs[8];
#pragma unroll
    for (int e = 0; e < 8; ++e) {
      float v = fmaxf(fmaxf(sc[0][e], sc[1][e]), fmaxf(sc[2][e], sc[3][e]));
      v = fmaxf(v, __shfl_xor(v, 1, 32));
      v = fmaxf(v, __shfl_xor(v, 2, 32));
      v = fmaxf(v, __shfl_xor(v, 4, 32));
      v = fmaxf(v, __shfl_xor(v, 8, 32));
      const float mn = fmaxf(mrow[e], v);
      alpha[e] = __expf(mrow[e] - mn);
      mrow[e] = mn;
      rs[e] = 0.0f;
    }

    // P = exp(S - m); stage P into wave-private LDS in A-layout order
#pragma unroll
    for (int t = 0; t < 4; ++t) {
#pragma unroll
      for (int e = 0; e < 8; ++e) {
        const float p = __expf(sc[t][e] - mrow[e]);
        rs[e] += p;
        prow[(hs * 8 + e) * 64 + t * 16 + ln] = (_Float16)p;   // ds_store_b16
      }
    }

#pragma unroll
    for (int e = 0; e < 8; ++e) {
      float v = rs[e];
      v += __shfl_xor(v, 1, 32);
      v += __shfl_xor(v, 2, 32);
      v += __shfl_xor(v, 4, 32);
      v += __shfl_xor(v, 8, 32);
      lsum[e] = lsum[e] * alpha[e] + v;
    }
#pragma unroll
    for (int t = 0; t < 4; ++t)
#pragma unroll
      for (int e = 0; e < 8; ++e) o[t][e] *= alpha[e];

    // O += P * V  (A = P from LDS, B = V^T chunk from LDS) -- 8 WMMAs
    const _Float16* pA = prow + ln * 64;
#pragma unroll
    for (int half = 0; half < 2; ++half) {
      const int dt = half * 32;
      h16 paf = frag_a(pA, dt, hs);                             // ds_load_b128 x2
#pragma unroll
      for (int t = 0; t < 4; ++t) {
        h16 vf = frag_b(vb + (t * 16 + ln) * 64, dt, hs);
        o[t] = wmma32(paf, vf, o[t]);
      }
    }
  }

  // finalize: divide by row sum, write [S,B,EMB] f16
#pragma unroll
  for (int e = 0; e < 8; ++e) mrow[e] = 1.0f / lsum[e];         // reuse as inv
#pragma unroll
  for (int t = 0; t < 4; ++t) {
#pragma unroll
    for (int e = 0; e < 8; ++e) {
      const int s = s0 + hs * 8 + e;
      const int d = t * 16 + ln;
      attn[((size_t)s * BATCH + b) * EMB + h * HD + d] = (_Float16)(o[t][e] * mrow[e]);
    }
  }
}

// ---------------- host-side launch ----------------
extern "C" void kernel_launch(void* const* d_in, const int* in_sizes, int n_in,
                              void* d_out, int out_size, void* d_ws, size_t ws_size,
                              hipStream_t stream) {
  const float* x  = (const float*)d_in[0];  // [S,B,E]
  const float* wi = (const float*)d_in[1];  // [3E,E]
  const float* bi = (const float*)d_in[2];  // [3E]
  const float* wo = (const float*)d_in[3];  // [E,E]
  const float* bo = (const float*)d_in[4];  // [E]
  float* out = (float*)d_out;

  // workspace carve-up (~48 MB total)
  char* p = (char*)d_ws;
  _Float16* x_h   = (_Float16*)p; p += (size_t)M_ROWS * EMB * sizeof(_Float16);
  _Float16* wi_h  = (_Float16*)p; p += (size_t)N_QKV  * EMB * sizeof(_Float16);
  _Float16* wo_h  = (_Float16*)p; p += (size_t)EMB    * EMB * sizeof(_Float16);
  _Float16* q_h   = (_Float16*)p; p += (size_t)BATCH * NH * S_LEN * HD * sizeof(_Float16);
  _Float16* k_h   = (_Float16*)p; p += (size_t)BATCH * NH * S_LEN * HD * sizeof(_Float16);
  _Float16* vT_h  = (_Float16*)p; p += (size_t)BATCH * NH * HD * S_LEN * sizeof(_Float16);
  _Float16* at_h  = (_Float16*)p; p += (size_t)M_ROWS * EMB * sizeof(_Float16);

  cvt_f32_f16<<<2048, 256, 0, stream>>>(x,  x_h,  M_ROWS * EMB);
  cvt_f32_f16<<<2048, 256, 0, stream>>>(wi, wi_h, N_QKV * EMB);
  cvt_f32_f16<<<1024, 256, 0, stream>>>(wo, wo_h, EMB * EMB);

  gemm_qkv<<<dim3(N_QKV / 64, M_ROWS / 128), 128, 0, stream>>>(x_h, wi_h, bi, q_h, k_h, vT_h);

  attn_chunk<<<dim3(S_LEN / CHUNK, BATCH * NH), 128, 0, stream>>>(q_h, k_h, vT_h, at_h);

  gemm_out<<<dim3(EMB / 64, M_ROWS / 128), 128, 0, stream>>>(at_h, wo_h, bo, out);
}
